// PointNet_61280593379643
// MI455X (gfx1250) — matrix operand order repaired and checked by reading, not compiled, not run on hardware
//
#include <hip/hip_runtime.h>
#include <hip/hip_bf16.h>

typedef __attribute__((ext_vector_type(16))) _Float16 v16h;
typedef __attribute__((ext_vector_type(8)))  float    v8f;
typedef __attribute__((ext_vector_type(4)))  unsigned int v4u;
typedef __attribute__((ext_vector_type(4)))  int v4i;

#define B_CLOUDS 16
#define N_PER    2048
#define KNN      16
#define NPTS_ALL (B_CLOUDS * N_PER)   // 32768

union V16H {
    v16h h;
    v4u  u[2];
    _Float16 e[16];
    unsigned int w[8];
};

__device__ __forceinline__ v8f wmma_f16(v16h a, v16h b, v8f c) {
    return __builtin_amdgcn_wmma_f32_16x16x32_f16(
        /*neg_a=*/false, a, /*neg_b=*/false, b,
        /*c_mod=*/(short)0, c, /*reuse_a=*/false, /*reuse_b=*/false);
}

__device__ __forceinline__ v16h load_btile(const unsigned int* __restrict__ base, int lane) {
    V16H r;
    const v4u* p = (const v4u*)(base + lane * 8);
    r.u[0] = p[0];
    r.u[1] = p[1];
    return r.h;
}

// ---------------------------------------------------------------------------
// Pack all weight matrices into WMMA B-layout f16 tiles, and zero the pool.
// B k order used by this code (16-bit, 32x16 per tile; A packed the same): lane L -> col N=L%16, element e -> K = e + (L>=16?16:0)
// Tiles: 0,1:W1a(Keff=6)  2,3:W1b  4,5:W2a rows0..31  6,7:W2a rows32..34(Keff=3)  8,9:W2b
// ---------------------------------------------------------------------------
__global__ void prep_kernel(const float* __restrict__ W1a, const float* __restrict__ W1b,
                            const float* __restrict__ W2a, const float* __restrict__ W2b,
                            unsigned int* __restrict__ wb, float* __restrict__ pool) {
    int t = threadIdx.x;
    if (t >= 512) {                        // zero pool[512]
        pool[t - 512] = 0.0f;
        return;
    }
    if (t >= 320) return;
    int tile = t >> 5;
    int lane = t & 31;
    const float* src; int r0; int keff;
    switch (tile) {
        case 0: case 1: src = W1a; r0 = 0;  keff = 6;  break;
        case 2: case 3: src = W1b; r0 = 0;  keff = 32; break;
        case 4: case 5: src = W2a; r0 = 0;  keff = 32; break;
        case 6: case 7: src = W2a; r0 = 32; keff = 3;  break;
        default:        src = W2b; r0 = 0;  keff = 32; break;
    }
    int n = (lane & 15) + ((tile & 1) ? 16 : 0);
    int kbase = (lane < 16) ? 0 : 16;
    unsigned int* out = wb + tile * 256 + lane * 8;
    #pragma unroll
    for (int v = 0; v < 8; ++v) {
        int k0 = kbase + 2 * v;
        int k1 = k0 + 1;
        float f0 = (k0 < keff) ? src[(r0 + k0) * 32 + n] : 0.0f;
        float f1 = (k1 < keff) ? src[(r0 + k1) * 32 + n] : 0.0f;
        union { _Float16 h[2]; unsigned int u; } pk;
        pk.h[0] = (_Float16)f0;
        pk.h[1] = (_Float16)f1;
        out[v] = pk.u;
    }
}

// ---------------------------------------------------------------------------
// KNN: one block = 256 query points of one cloud. Cloud (24 KB contiguous) is
// staged into LDS via the CDNA5 async global->LDS path (ASYNCcnt-tracked,
// no VGPR round-trip), then each thread keeps a register-resident sorted
// top-16 over broadcast LDS reads.
// ---------------------------------------------------------------------------
__global__ __launch_bounds__(256) void knn_kernel(const float* __restrict__ pos,
                                                  int* __restrict__ idxout) {
    __shared__ __align__(16) float sp[N_PER * 3];   // 24 KB packed xyz
    int b = blockIdx.x >> 3;
    int qbase = (blockIdx.x & 7) * 256;
    const float* pb = pos + (size_t)b * N_PER * 3;

#if __has_builtin(__builtin_amdgcn_global_load_async_to_lds_b128)
    {
        typedef __attribute__((address_space(1))) v4i* gv4i_p;
        typedef __attribute__((address_space(3))) v4i* lv4i_p;
        __attribute__((address_space(1))) char* g =
            (__attribute__((address_space(1))) char*)pb;
        __attribute__((address_space(3))) char* l =
            (__attribute__((address_space(3))) char*)sp;
        int t = threadIdx.x;
        #pragma unroll
        for (int r = 0; r < 6; ++r) {           // 1536 x b128 = 24 KB
            int off = (t + r * 256) * 16;
            __builtin_amdgcn_global_load_async_to_lds_b128(
                (gv4i_p)(g + off), (lv4i_p)(l + off), 0, 0);
        }
#if __has_builtin(__builtin_amdgcn_s_wait_asynccnt)
        __builtin_amdgcn_s_wait_asynccnt(0);
#else
        asm volatile("s_wait_asynccnt 0" ::: "memory");
#endif
    }
#else
    for (int t = threadIdx.x; t < N_PER * 3 / 4; t += 256)
        ((float4*)sp)[t] = ((const float4*)pb)[t];
#endif
    __syncthreads();

    int q = qbase + threadIdx.x;
    float px = sp[q * 3], py = sp[q * 3 + 1], pz = sp[q * 3 + 2];
    float bd[KNN]; int bi[KNN];
    #pragma unroll
    for (int k = 0; k < KNN; ++k) { bd[k] = 3.4e38f; bi[k] = 0; }

    for (int j = 0; j < N_PER; ++j) {
        float sx = sp[j * 3], sy = sp[j * 3 + 1], sz = sp[j * 3 + 2];
        float dx = sx - px, dy = sy - py, dz = sz - pz;
        float d2 = dx * dx + dy * dy + dz * dz;
        if (d2 < bd[KNN - 1]) {
            float cd = d2; int ci = j;
            #pragma unroll
            for (int k = 0; k < KNN; ++k) {
                bool sw = cd < bd[k];
                float td = bd[k]; int ti = bi[k];
                if (sw) { bd[k] = cd; bi[k] = ci; cd = td; ci = ti; }
            }
        }
    }
    int* o = idxout + (size_t)(b * N_PER + q) * KNN;
    #pragma unroll
    for (int k = 0; k < KNN; ++k) o[k] = bi[k];
}

// ---------------------------------------------------------------------------
// Layer 1: one wave per point. A tile rows = the point's 16 neighbors,
// features [pos_j(3), pos_j - pos_i(3)] zero-padded to K=32.
// GEMM1 (6->32): 2 WMMA; ReLU+bias; LDS transpose D->A; GEMM2 (32->32): 2 WMMA;
// max over M (neighbors) + bias + ReLU -> h1 (f16).
// ---------------------------------------------------------------------------
__global__ __launch_bounds__(256) void layer1_kernel(
        const float* __restrict__ pos, const int* __restrict__ knn,
        const unsigned int* __restrict__ wb,
        const float* __restrict__ b1a, const float* __restrict__ b1b,
        _Float16* __restrict__ h1out) {
    __shared__ __align__(16) _Float16 hid[8][16][32];   // 8 KB
    int lane = threadIdx.x & 31;
    int w    = threadIdx.x >> 5;
    int i    = blockIdx.x * 8 + w;        // point id 0..32767
    int b    = i >> 11;
    int li   = i & (N_PER - 1);
    int m    = lane & 15;

    const float* pb = pos + (size_t)b * N_PER * 3;
    int jj = knn[(size_t)i * KNN + m];
    float pjx = pb[jj * 3], pjy = pb[jj * 3 + 1], pjz = pb[jj * 3 + 2];
    float pix = pb[li * 3], piy = pb[li * 3 + 1], piz = pb[li * 3 + 2];

    V16H a;
    #pragma unroll
    for (int e = 0; e < 8; ++e) a.w[e] = 0u;
    if (lane < 16) {   // A layout: lanes 0-15 hold K=0..7 in elements 0..7
        a.e[0] = (_Float16)pjx; a.e[1] = (_Float16)pjy; a.e[2] = (_Float16)pjz;
        a.e[3] = (_Float16)(pjx - pix);
        a.e[4] = (_Float16)(pjy - piy);
        a.e[5] = (_Float16)(pjz - piz);
    }

    v16h B0 = load_btile(wb + 0 * 256, lane);
    v16h B1 = load_btile(wb + 1 * 256, lane);
    v8f c0 = {}, c1 = {};
    c0 = wmma_f16(a.h, B0, c0);
    c1 = wmma_f16(a.h, B1, c1);

    // bias + ReLU, store transposed into LDS (D layout -> [M][K])
    int n = lane & 15;
    float ba0 = b1a[n], ba1 = b1a[n + 16];
    int mrow = (lane < 16) ? 0 : 8;
    #pragma unroll
    for (int v = 0; v < 8; ++v) {
        float x0 = c0[v] + ba0; x0 = x0 > 0.0f ? x0 : 0.0f;
        float x1 = c1[v] + ba1; x1 = x1 > 0.0f ? x1 : 0.0f;
        hid[w][mrow + v][n]      = (_Float16)x0;
        hid[w][mrow + v][n + 16] = (_Float16)x1;
    }
    __syncthreads();

    // read back in A layout: row m, channels kb..kb+15
    int kb = (lane < 16) ? 0 : 16;
    V16H a2;
    a2.u[0] = *(const v4u*)&hid[w][m][kb];
    a2.u[1] = *(const v4u*)&hid[w][m][kb + 8];

    v16h B2 = load_btile(wb + 2 * 256, lane);
    v16h B3 = load_btile(wb + 3 * 256, lane);
    v8f d0 = {}, d1 = {};
    d0 = wmma_f16(a2.h, B2, d0);
    d1 = wmma_f16(a2.h, B3, d1);

    // max over neighbors (M dim): per-lane over 8 VGPRs, then across lane halves
    float m0 = d0[0], m1 = d1[0];
    #pragma unroll
    for (int v = 1; v < 8; ++v) { m0 = fmaxf(m0, d0[v]); m1 = fmaxf(m1, d1[v]); }
    m0 = fmaxf(m0, __shfl_xor(m0, 16, 32));
    m1 = fmaxf(m1, __shfl_xor(m1, 16, 32));

    float vb; int ch;
    if (lane < 16) { vb = m0 + b1b[n];      ch = n; }
    else           { vb = m1 + b1b[n + 16]; ch = n + 16; }
    vb = vb > 0.0f ? vb : 0.0f;
    h1out[(size_t)i * 32 + ch] = (_Float16)vb;
}

// ---------------------------------------------------------------------------
// Layer 2: features = [h1_j(32), pos_j - pos_i(3)] -> K=64 as two K-chunks.
// Gathered h1 rows load directly into A layout with two b128 loads per lane.
// Per-point results pooled block-locally in LDS (ds_max), then one 32-lane
// global atomicMax flush per block (values >= 0 so uint bit order == float).
// ---------------------------------------------------------------------------
__global__ __launch_bounds__(256) void layer2_kernel(
        const float* __restrict__ pos, const int* __restrict__ knn,
        const _Float16* __restrict__ h1,
        const unsigned int* __restrict__ wb,
        const float* __restrict__ b2a, const float* __restrict__ b2b,
        unsigned int* __restrict__ pool) {
    __shared__ __align__(16) _Float16 hid[8][16][32];
    __shared__ unsigned int bpool[32];
    if (threadIdx.x < 32) bpool[threadIdx.x] = 0u;   // completes before mid-kernel barrier

    int lane = threadIdx.x & 31;
    int w    = threadIdx.x >> 5;
    int i    = blockIdx.x * 8 + w;
    int b    = i >> 11;                    // uniform across block (2048 % 8 == 0)
    int li   = i & (N_PER - 1);
    int m    = lane & 15;

    const float* pb = pos + (size_t)b * N_PER * 3;
    int jj = knn[(size_t)i * KNN + m];
    int gj = (b << 11) + jj;

    // A chunk0: h1_j channels in A layout (contiguous 8-channel groups)
    int kb = (lane < 16) ? 0 : 16;
    V16H a0;
    {
        const v4u* hp = (const v4u*)(h1 + (size_t)gj * 32);
        a0.u[0] = hp[kb >> 3];
        a0.u[1] = hp[(kb >> 3) + 1];
    }
    // A chunk1: spatial (K'=0..2), rest zero
    V16H a1;
    #pragma unroll
    for (int e = 0; e < 8; ++e) a1.w[e] = 0u;
    if (lane < 16) {
        float pjx = pb[jj * 3], pjy = pb[jj * 3 + 1], pjz = pb[jj * 3 + 2];
        float pix = pb[li * 3], piy = pb[li * 3 + 1], piz = pb[li * 3 + 2];
        a1.e[0] = (_Float16)(pjx - pix);
        a1.e[1] = (_Float16)(pjy - piy);
        a1.e[2] = (_Float16)(pjz - piz);
    }

    v16h B4 = load_btile(wb + 4 * 256, lane);
    v16h B5 = load_btile(wb + 5 * 256, lane);
    v16h B6 = load_btile(wb + 6 * 256, lane);
    v16h B7 = load_btile(wb + 7 * 256, lane);
    v8f c0 = {}, c1 = {};
    c0 = wmma_f16(a0.h, B4, c0);
    c0 = wmma_f16(a1.h, B6, c0);
    c1 = wmma_f16(a0.h, B5, c1);
    c1 = wmma_f16(a1.h, B7, c1);

    int n = lane & 15;
    float ba0 = b2a[n], ba1 = b2a[n + 16];
    int mrow = (lane < 16) ? 0 : 8;
    #pragma unroll
    for (int v = 0; v < 8; ++v) {
        float x0 = c0[v] + ba0; x0 = x0 > 0.0f ? x0 : 0.0f;
        float x1 = c1[v] + ba1; x1 = x1 > 0.0f ? x1 : 0.0f;
        hid[w][mrow + v][n]      = (_Float16)x0;
        hid[w][mrow + v][n + 16] = (_Float16)x1;
    }
    __syncthreads();

    V16H a2;
    a2.u[0] = *(const v4u*)&hid[w][m][kb];
    a2.u[1] = *(const v4u*)&hid[w][m][kb + 8];

    v16h B8 = load_btile(wb + 8 * 256, lane);
    v16h B9 = load_btile(wb + 9 * 256, lane);
    v8f d0 = {}, d1 = {};
    d0 = wmma_f16(a2.h, B8, d0);
    d1 = wmma_f16(a2.h, B9, d1);

    float m0 = d0[0], m1 = d1[0];
    #pragma unroll
    for (int v = 1; v < 8; ++v) { m0 = fmaxf(m0, d0[v]); m1 = fmaxf(m1, d1[v]); }
    m0 = fmaxf(m0, __shfl_xor(m0, 16, 32));
    m1 = fmaxf(m1, __shfl_xor(m1, 16, 32));

    float vb; int ch;
    if (lane < 16) { vb = m0 + b2b[n];      ch = n; }
    else           { vb = m1 + b2b[n + 16]; ch = n + 16; }
    vb = vb > 0.0f ? vb : 0.0f;

    // block-local LDS max reduction, then one global flush per block
    atomicMax(&bpool[ch], __float_as_uint(vb));
    __syncthreads();
    if (threadIdx.x < 32)
        atomicMax(pool + b * 32 + threadIdx.x, bpool[threadIdx.x]);
}

// ---------------------------------------------------------------------------
// Classifier: out[b,c] = pool[b,:] @ Wc + bc   (16x10, trivial)
// ---------------------------------------------------------------------------
__global__ void classifier_kernel(const unsigned int* __restrict__ pool,
                                  const float* __restrict__ Wc,
                                  const float* __restrict__ bc,
                                  float* __restrict__ out) {
    int t = threadIdx.x;
    if (t >= 160) return;
    int b = t / 10, c = t % 10;
    float s = bc[c];
    #pragma unroll
    for (int n = 0; n < 32; ++n)
        s += __uint_as_float(pool[b * 32 + n]) * Wc[n * 10 + c];
    out[b * 10 + c] = s;
}

extern "C" void kernel_launch(void* const* d_in, const int* in_sizes, int n_in,
                              void* d_out, int out_size, void* d_ws, size_t ws_size,
                              hipStream_t stream) {
    const float* pos = (const float*)d_in[0];
    // d_in[1] = batch (int64): unused — graphs are equal-sized and sorted
    const float* W1a = (const float*)d_in[2];
    const float* b1a = (const float*)d_in[3];
    const float* W1b = (const float*)d_in[4];
    const float* b1b = (const float*)d_in[5];
    const float* W2a = (const float*)d_in[6];
    const float* b2a = (const float*)d_in[7];
    const float* W2b = (const float*)d_in[8];
    const float* b2b = (const float*)d_in[9];
    const float* Wc  = (const float*)d_in[10];
    const float* bc  = (const float*)d_in[11];

    char* ws = (char*)d_ws;
    int*          idx  = (int*)(ws);                                    // 2 MB
    _Float16*     h1   = (_Float16*)(ws + (size_t)2 * 1024 * 1024);     // 2 MB
    unsigned int* wb   = (unsigned int*)(ws + (size_t)4 * 1024 * 1024); // 10 KB
    unsigned int* pool = (unsigned int*)(ws + (size_t)4 * 1024 * 1024 + 16384); // 2 KB

    prep_kernel<<<1, 1024, 0, stream>>>(W1a, W1b, W2a, W2b, wb, (float*)pool);
    knn_kernel<<<B_CLOUDS * 8, 256, 0, stream>>>(pos, idx);
    layer1_kernel<<<NPTS_ALL / 8, 256, 0, stream>>>(pos, idx, wb, b1a, b1b, h1);
    layer2_kernel<<<NPTS_ALL / 8, 256, 0, stream>>>(pos, idx, h1, wb, b2a, b2b, pool);
    classifier_kernel<<<1, 256, 0, stream>>>(pool, Wc, bc, (float*)d_out);
}
